// DiffusionGraphConv_71786083385975
// MI455X (gfx1250) — matrix operand order repaired and checked by reading, not compiled, not run on hardware
//
#include <hip/hip_runtime.h>

typedef __attribute__((ext_vector_type(16))) __bf16 v16bf;
typedef __attribute__((ext_vector_type(8)))  float  v8f;

#define N_NODES 10000
#define BATCH   16
#define FDIM    128            // INPUT_DIM + HID_DIM
#define ROWLEN  2048           // BATCH * FDIM
#define OUTDIM  128
#define NMAT    3
#define KSTEPS  12             // 384 / 32
#define NTILES  8              // OUTDIM / 16
#define XELEMS  ((size_t)N_NODES * ROWLEN)             // 20,480,000 floats per plane
#define WFRAG_USHORTS (KSTEPS * NTILES * 2 * 32 * 16)  // 98304 (196608 bytes)

// ---- bf16 helpers (round-to-nearest-even, bit-level) ----
__device__ __forceinline__ unsigned short f2bf_u(float f) {
    unsigned u = __builtin_bit_cast(unsigned, f);
    u += 0x7fffu + ((u >> 16) & 1u);
    return (unsigned short)(u >> 16);
}
__device__ __forceinline__ float bfu2f(unsigned short s) {
    unsigned u = ((unsigned)s) << 16;
    return __builtin_bit_cast(float, u);
}

// Build hi/lo bf16 A-fragments from 16 fp32 values (4 float4 quads)
__device__ __forceinline__ void split16(float4 q0, float4 q1, float4 q2, float4 q3,
                                        v16bf& hi, v16bf& lo) {
    float av[16] = { q0.x,q0.y,q0.z,q0.w, q1.x,q1.y,q1.z,q1.w,
                     q2.x,q2.y,q2.z,q2.w, q3.x,q3.y,q3.z,q3.w };
    #pragma unroll
    for (int j = 0; j < 16; ++j) {
        unsigned short h = f2bf_u(av[j]);
        hi[j] = __builtin_bit_cast(__bf16, h);
        lo[j] = __builtin_bit_cast(__bf16, f2bf_u(av[j] - bfu2f(h)));
    }
}

// ---------------------------------------------------------------------------
// Kernel 1: x0[n][b][f] = concat(inputs, state);   layout [N, B, F], float4-wide
// ---------------------------------------------------------------------------
__global__ void __launch_bounds__(256) build_x0_kernel(
    const float4* __restrict__ inputs4, const float4* __restrict__ state4,
    float4* __restrict__ x04)
{
    size_t t = (size_t)blockIdx.x * 256 + threadIdx.x;   // 0 .. XELEMS/4-1
    int    f4 = (int)(t & 31);                           // float4 index within 128-f row
    size_t nb = t >> 5;
    int    b  = (int)(nb & 15);
    size_t n  = nb >> 4;
    float4 v = (f4 < 16) ? inputs4[(size_t)b * (N_NODES * 16) + n * 16 + f4]
                         : state4 [(size_t)b * (N_NODES * 16) + n * 16 + (f4 - 16)];
    x04[t] = v;
}

// ---------------------------------------------------------------------------
// Kernel 2/3: out[r,:] = alpha * sum_e(val[e] * z[col[e],:]) + beta * aux[r,:]
// One block per row; rows[] sorted -> binary search; float4-wide gathers.
// ---------------------------------------------------------------------------
__global__ void __launch_bounds__(256) spmm_kernel(
    const float* __restrict__ z,
    const int*   __restrict__ rows, const int* __restrict__ cols,
    const float* __restrict__ vals, int nnz,
    const float* __restrict__ aux, float alpha, float beta,
    float* __restrict__ outp)
{
    int r = blockIdx.x;
    int lo = 0, hi = nnz;
    while (lo < hi) { int mid = (lo + hi) >> 1; if (rows[mid] < r) lo = mid + 1; else hi = mid; }
    int s = lo;
    int lo2 = s, hi2 = nnz;
    while (lo2 < hi2) { int mid = (lo2 + hi2) >> 1; if (rows[mid] < r + 1) lo2 = mid + 1; else hi2 = mid; }
    int e = lo2;

    int cb = threadIdx.x * 8;                       // 8 consecutive columns per thread
    float4 acc0 = {0.f,0.f,0.f,0.f}, acc1 = {0.f,0.f,0.f,0.f};
    for (int i = s; i < e; ++i) {
        float v = vals[i];
        const float4* zr = (const float4*)(z + (size_t)cols[i] * ROWLEN + cb);
        float4 z0 = zr[0], z1 = zr[1];
        acc0.x += v * z0.x; acc0.y += v * z0.y; acc0.z += v * z0.z; acc0.w += v * z0.w;
        acc1.x += v * z1.x; acc1.y += v * z1.y; acc1.z += v * z1.z; acc1.w += v * z1.w;
    }
    float4* o = (float4*)(outp + (size_t)r * ROWLEN + cb);
    if (beta != 0.f) {
        const float4* a = (const float4*)(aux + (size_t)r * ROWLEN + cb);
        float4 a0 = a[0], a1 = a[1];
        float4 r0 = { alpha*acc0.x + beta*a0.x, alpha*acc0.y + beta*a0.y,
                      alpha*acc0.z + beta*a0.z, alpha*acc0.w + beta*a0.w };
        float4 r1 = { alpha*acc1.x + beta*a1.x, alpha*acc1.y + beta*a1.y,
                      alpha*acc1.z + beta*a1.z, alpha*acc1.w + beta*a1.w };
        o[0] = r0; o[1] = r1;
    } else {
        float4 r0 = { alpha*acc0.x, alpha*acc0.y, alpha*acc0.z, alpha*acc0.w };
        float4 r1 = { alpha*acc1.x, alpha*acc1.y, alpha*acc1.z, alpha*acc1.w };
        o[0] = r0; o[1] = r1;
    }
}

// ---------------------------------------------------------------------------
// Kernel 4a: pre-split weight[384,128] (row k_orig = f*3 + m) into bf16 hi/lo
// B-fragments in exact WMMA lane order under permuted K = m*128 + f.
// B 16-bit 32x16 layout: lane L -> col L%16; K {o..o+7, o+16..o+23}, o=(L<16?0:8).
// Fragment offset(t, ct, plane, lane, e) = ((((t*8+ct)*2+plane)*32)+lane)*16 + e
// ---------------------------------------------------------------------------
__global__ void __launch_bounds__(256) prep_wfrag_kernel(
    const float* __restrict__ W, unsigned short* __restrict__ Wf)
{
    int idx = blockIdx.x * 256 + threadIdx.x;
    if (idx >= KSTEPS * NTILES * 32 * 16) return;   // 49152
    int e    =  idx        & 15;
    int lane = (idx >>  4) & 31;
    int ct   = (idx >>  9) & 7;
    int t    =  idx >> 12;                           // 0..11
    int klocal = ((lane >> 4) << 3) + (e & 7) + ((e >> 3) << 4);   // 0..31
    int m  = t >> 2;
    int f  = ((t & 3) << 5) + klocal;
    int korig = f * NMAT + m;
    int col   = ct * 16 + (lane & 15);
    float w = W[(size_t)korig * OUTDIM + col];
    unsigned short hi = f2bf_u(w);
    unsigned short lo = f2bf_u(w - bfu2f(hi));
    int base = ((((t * NTILES + ct) * 2) * 32) + lane) * 16 + e;
    Wf[base]           = hi;     // plane 0
    Wf[base + 32 * 16] = lo;     // plane 1
}

// ---------------------------------------------------------------------------
// Kernel 5: GEMM [160000 x 384] @ [384 x 128] via v_wmma_f32_16x16x32_bf16,
// 3-term bf16 hi/lo split. Block = 8 waves, 625 blocks (one per 16-row n-tile).
// Each wave owns TWO M-tiles (batches b=wave and b=wave+8) so every B-fragment
// read from LDS feeds 6 WMMAs. Weight fragments staged into LDS with
// GLOBAL_LOAD_ASYNC_TO_LDS_B128 (ASYNCcnt), bypassing the register file.
// A 16-bit 16x32 layout: lane L -> row n0+(L&15); K {o..o+7, o+16..o+23}, o=(L<16?0:8).
// ---------------------------------------------------------------------------
__global__ void __launch_bounds__(256, 1) gemm_wmma_kernel(
    const float* __restrict__ x0, const float* __restrict__ x1, const float* __restrict__ x2,
    const unsigned short* __restrict__ Wf, const float* __restrict__ biases,
    float* __restrict__ out)
{
    extern __shared__ unsigned short sW[];           // 196608 bytes
    {   // async global -> LDS staging (GVS mode: vdst = LDS addr, vaddr = offset, saddr = base)
        unsigned lbase = (unsigned)(unsigned long long)(void*)sW;  // low 32 bits = LDS byte addr
        for (int i = threadIdx.x; i < WFRAG_USHORTS / 8; i += 256) {
            unsigned loff = lbase + (unsigned)(i * 16);
            unsigned goff = (unsigned)(i * 16);
            asm volatile("global_load_async_to_lds_b128 %0, %1, %2"
                         :
                         : "v"(loff), "v"(goff), "s"(Wf)
                         : "memory");
        }
        asm volatile("s_wait_asynccnt 0" ::: "memory");
    }
    __syncthreads();

    int lane  = threadIdx.x & 31;
    int wave  = threadIdx.x >> 5;                    // 0..7
    int n0    = blockIdx.x << 4;                     // 16-row tile base (625 blocks)
    int b0    = wave;                                // first batch
    int b1    = wave + 8;                            // second batch
    int mrow  = lane & 15;
    int khalf = lane >> 4;                           // 0 or 1 (K offset +8)
    size_t arow0 = ((size_t)(n0 + mrow) * BATCH + b0) * FDIM;
    size_t arow1 = ((size_t)(n0 + mrow) * BATCH + b1) * FDIM;

    v8f c0[NTILES] = {};
    v8f c1[NTILES] = {};

    for (int t = 0; t < KSTEPS; ++t) {
        const float* xm = (t < 4) ? x0 : ((t < 8) ? x1 : x2);
        int koff = ((t & 3) << 5) + khalf * 8;
        const float4* p0 = (const float4*)(xm + arow0 + koff);
        const float4* p1 = (const float4*)(xm + arow1 + koff);
        v16bf a0h, a0l, a1h, a1l;
        split16(p0[0], p0[1], p0[4], p0[5], a0h, a0l);
        split16(p1[0], p1[1], p1[4], p1[5], a1h, a1l);
        #pragma unroll
        for (int ct = 0; ct < NTILES; ++ct) {
            const unsigned short* bp =
                sW + (((((t * NTILES + ct) * 2) * 32) + lane) << 4);
            v16bf bh = *(const v16bf*)bp;             // hi plane
            v16bf bl = *(const v16bf*)(bp + 32 * 16); // lo plane
            c0[ct] = __builtin_amdgcn_wmma_f32_16x16x32_bf16(
                         false, a0h, false, bh, (short)0, c0[ct], false, false);
            c0[ct] = __builtin_amdgcn_wmma_f32_16x16x32_bf16(
                         false, a0h, false, bl, (short)0, c0[ct], false, false);
            c0[ct] = __builtin_amdgcn_wmma_f32_16x16x32_bf16(
                         false, a0l, false, bh, (short)0, c0[ct], false, false);
            c1[ct] = __builtin_amdgcn_wmma_f32_16x16x32_bf16(
                         false, a1h, false, bh, (short)0, c1[ct], false, false);
            c1[ct] = __builtin_amdgcn_wmma_f32_16x16x32_bf16(
                         false, a1h, false, bl, (short)0, c1[ct], false, false);
            c1[ct] = __builtin_amdgcn_wmma_f32_16x16x32_bf16(
                         false, a1l, false, bh, (short)0, c1[ct], false, false);
        }
    }

    // Epilogue: C layout -> VGPR i: rows (n0+i | n0+8+i), col = ct*16 + (lane&15)
    size_t obase0 = (size_t)b0 * ((size_t)N_NODES * OUTDIM);
    size_t obase1 = (size_t)b1 * ((size_t)N_NODES * OUTDIM);
    #pragma unroll
    for (int ct = 0; ct < NTILES; ++ct) {
        int col = ct * 16 + (lane & 15);
        float bias = biases[col];
        #pragma unroll
        for (int i = 0; i < 8; ++i) {
            size_t roff = (size_t)(n0 + i + khalf * 8) * OUTDIM + col;
            out[obase0 + roff] = c0[ct][i] + bias;
            out[obase1 + roff] = c1[ct][i] + bias;
        }
    }
}

// ---------------------------------------------------------------------------
extern "C" void kernel_launch(void* const* d_in, const int* in_sizes, int n_in,
                              void* d_out, int out_size, void* d_ws, size_t ws_size,
                              hipStream_t stream)
{
    (void)n_in; (void)out_size; (void)ws_size;
    const float* inputs = (const float*)d_in[0];
    const float* state  = (const float*)d_in[1];
    const int*   rows   = (const int*)  d_in[2];
    const int*   cols   = (const int*)  d_in[3];
    const float* vals   = (const float*)d_in[4];
    const float* weight = (const float*)d_in[5];
    const float* biases = (const float*)d_in[6];
    float* out = (float*)d_out;
    int nnz = in_sizes[2];

    float* x0 = (float*)d_ws;
    float* x1 = x0 + XELEMS;
    float* x2 = x1 + XELEMS;
    unsigned short* Wf = (unsigned short*)(x2 + XELEMS);

    build_x0_kernel<<<(unsigned)(XELEMS / 4 / 256), 256, 0, stream>>>(
        (const float4*)inputs, (const float4*)state, (float4*)x0);
    prep_wfrag_kernel<<<(KSTEPS * NTILES * 32 * 16 + 255) / 256, 256, 0, stream>>>(weight, Wf);
    // x1 = S @ x0
    spmm_kernel<<<N_NODES, 256, 0, stream>>>(x0, rows, cols, vals, nnz, nullptr, 1.f, 0.f, x1);
    // x2 = 2 * (S @ x1) - x0
    spmm_kernel<<<N_NODES, 256, 0, stream>>>(x1, rows, cols, vals, nnz, x0, 2.f, -1.f, x2);
    // out = [x0|x1|x2] @ W + bias  (3x bf16-split WMMA)
    gemm_wmma_kernel<<<625, 256, WFRAG_USHORTS * sizeof(unsigned short), stream>>>(
        x0, x1, x2, Wf, biases, out);
}
